// BiAttention_65128884076946
// MI455X (gfx1250) — compile-verified, matrix operand order checked by reading, not run on hardware
//
#include <hip/hip_runtime.h>
#include <hip/hip_bf16.h>

// ---------------------------------------------------------------------------
// BiAttention (trilinear attention) for MI455X / gfx1250.
// B=16, L=1024, D=256.  bf16 WMMA (v_wmma_f32_16x16x32_bf16) for all GEMMs,
// TDM (tensor_load_to_lds) double-buffered LDS staging, f32 accumulation,
// f32 softmax.
// ---------------------------------------------------------------------------

#define BB 16
#define LL 1024
#define DD 256

// Padded LDS row: 32 bf16 (64B) + 16B pad = 80B  -> conflict-free b128 reads.
#define LDS_ROW_B 80

typedef __attribute__((ext_vector_type(16))) __bf16 v16bf;
typedef __attribute__((ext_vector_type(8)))  __bf16 v8bf;
typedef __attribute__((ext_vector_type(8)))  float  v8f;
typedef __attribute__((ext_vector_type(4)))  unsigned int v4u;
typedef __attribute__((ext_vector_type(8)))  int    v8i;
typedef __attribute__((ext_vector_type(4)))  int    v4i;

// ---------------------------------------------------------------------------
// TDM: async DMA of a [nrows x 32] bf16 tile (row stride = strideElems) from
// global memory into LDS at byte offset ldsOff, with TDM-applied padding of
// 16B after every 64B so LDS rows land at LDS_ROW_B stride.
// D# group0: count=1, lds_addr, global_addr(57b), type=2.
// D# group1: data_size=2B, pad_enable, pad_interval=16DW, pad_amount=4DW,
//            tensor_dim = tile_dim (tiles are interior), dim0 stride.
// Tracked by TENSORcnt.  This toolchain uses the 6-arg builtin form:
// (v4u g0, v8i g1, v4i g2, v4i g3, v8i extra, i32 cpol).
// ---------------------------------------------------------------------------
__device__ inline void tdm_load_tile(unsigned ldsOff, const __bf16* gptr,
                                     int nrows, long strideElems) {
    unsigned long long ga = (unsigned long long)gptr;
    v4u g0;
    g0[0] = 1u;                                             // count=1 (valid)
    g0[1] = ldsOff;                                         // lds_addr
    g0[2] = (unsigned)(ga & 0xFFFFFFFFull);                 // global_addr lo
    g0[3] = (unsigned)((ga >> 32) & 0x01FFFFFFull) | (2u << 30);  // hi | type=2
    v8i g1;
    g1[0] = (1 << 16) | (1 << 20) | (3 << 22) | (3 << 25);  // 2B, pad 4DW/16DW
    g1[1] = (32 & 0xFFFF) << 16;                            // tensor_dim0 = 32
    g1[2] = (nrows & 0xFFFF) << 16;                         // tensor_dim1 lo
    g1[3] = (32 << 16);                                     // tile_dim0 = 32
    g1[4] = (nrows & 0xFFFF);                               // tile_dim1
    g1[5] = (int)(strideElems & 0xFFFFFFFFl);               // dim0 stride lo
    g1[6] = (int)((strideElems >> 32) & 0xFFFFl);           // dim0 stride hi
    g1[7] = 0;
    v4i gz4 = {0, 0, 0, 0};
    v8i gz8 = {0, 0, 0, 0, 0, 0, 0, 0};
    __builtin_amdgcn_tensor_load_to_lds(g0, g1, gz4, gz4, gz8, 0);
}

// ---------------------------------------------------------------------------
// Fragment loader from padded LDS: 16-bit A/B operand, 16x32 tile.
// Lane = 16*h + m holds row m, K-chunks [8h,8h+8) and [16+8h,24+8h):
// two ds_load_b128 per lane, bank-conflict free at 80B row stride.
// ---------------------------------------------------------------------------
__device__ inline v16bf lds_frag(const unsigned char* buf, int row0, int lane) {
    const int m = lane & 15;
    const int h = lane >> 4;
    const unsigned char* p = buf + (row0 + m) * LDS_ROW_B + 16 * h;
    v8bf lo = *reinterpret_cast<const v8bf*>(p);
    v8bf hi = *reinterpret_cast<const v8bf*>(p + 32);
    return __builtin_shufflevector(lo, hi, 0, 1, 2, 3, 4, 5, 6, 7,
                                   8, 9, 10, 11, 12, 13, 14, 15);
}

// ---------------------------------------------------------------------------
// Generic batched NT GEMM:  C[b, i, n] = sum_k A[b, i, k] * Bm[b, n, k]
// Block: 256 threads = 8 waves (4Mx2N); block tile 128x64; wave tile 32x32
// (2x2 WMMA).  K staged in 32-wide chunks by TDM, double buffered.
// EPI 0: f32 out.  EPI 1: f32 out + s0[row] + s1[col] + bias.  EPI 2: bf16 out.
// ---------------------------------------------------------------------------
template <int EPI>
__global__ __launch_bounds__(256) void gemm_nt(
    const __bf16* __restrict__ A, int lda, long sA,
    const __bf16* __restrict__ Bm, int ldb, long sB,
    float* __restrict__ Cf, __bf16* __restrict__ Cb, int ldc, long sC,
    int K,
    const float* __restrict__ s0v, const float* __restrict__ s1v,
    const float* __restrict__ biasPtr) {
    __shared__ __align__(16) unsigned char ldsA[2][128 * LDS_ROW_B];  // 2x10KB
    __shared__ __align__(16) unsigned char ldsB[2][64 * LDS_ROW_B];   // 2x 5KB

    const int b = blockIdx.z;
    const __bf16* Ab = A + (long)b * sA;
    const __bf16* Bb = Bm + (long)b * sB;

    const int lane = threadIdx.x & 31;
    const int w    = threadIdx.x >> 5;
    const int wm   = w & 3;   // 0..3
    const int wn   = w >> 2;  // 0..1
    const int rowBlock = blockIdx.y * 128;
    const int colBlock = blockIdx.x * 64;
    const int rowBase  = rowBlock + wm * 32;
    const int colBase  = colBlock + wn * 32;

    const unsigned offA0 = (unsigned)(unsigned long long)(const void*)&ldsA[0][0];
    const unsigned offA1 = (unsigned)(unsigned long long)(const void*)&ldsA[1][0];
    const unsigned offB0 = (unsigned)(unsigned long long)(const void*)&ldsB[0][0];
    const unsigned offB1 = (unsigned)(unsigned long long)(const void*)&ldsB[1][0];

    const int nk = K >> 5;

    // Prologue: wave 0 DMAs tile 0 and waits on TENSORcnt.
    if (w == 0) {
        tdm_load_tile(offA0, Ab + (long)rowBlock * lda, 128, lda);
        tdm_load_tile(offB0, Bb + (long)colBlock * ldb, 64, ldb);
        __builtin_amdgcn_s_wait_tensorcnt(0);
    }
    __syncthreads();

    v8f acc[2][2] = {};

    for (int kk = 0; kk < nk; ++kk) {
        const int cur = kk & 1;
        const bool more = (kk + 1) < nk;

        // Issue next tile's DMA before computing the current one.
        if (w == 0 && more) {
            const int k1 = (kk + 1) << 5;
            tdm_load_tile(cur ? offA0 : offA1,
                          Ab + (long)rowBlock * lda + k1, 128, lda);
            tdm_load_tile(cur ? offB0 : offB1,
                          Bb + (long)colBlock * ldb + k1, 64, ldb);
        }

        const unsigned char* bufA = ldsA[cur];
        const unsigned char* bufB = ldsB[cur];
        v16bf a0 = lds_frag(bufA, wm * 32,      lane);
        v16bf a1 = lds_frag(bufA, wm * 32 + 16, lane);
        v16bf b0 = lds_frag(bufB, wn * 32,      lane);
        v16bf b1 = lds_frag(bufB, wn * 32 + 16, lane);

        acc[0][0] = __builtin_amdgcn_wmma_f32_16x16x32_bf16(
            false, a0, false, b0, (short)0, acc[0][0], false, false);
        acc[0][1] = __builtin_amdgcn_wmma_f32_16x16x32_bf16(
            false, a0, false, b1, (short)0, acc[0][1], false, false);
        acc[1][0] = __builtin_amdgcn_wmma_f32_16x16x32_bf16(
            false, a1, false, b0, (short)0, acc[1][0], false, false);
        acc[1][1] = __builtin_amdgcn_wmma_f32_16x16x32_bf16(
            false, a1, false, b1, (short)0, acc[1][1], false, false);

        if (w == 0 && more) __builtin_amdgcn_s_wait_tensorcnt(0);
        __syncthreads();
    }

    // C/D layout: lane = 16*h + n; VGPR r holds row (r + 8h) of the 16x16 tile.
    const int n = lane & 15;
    const int h = lane >> 4;
    float biasv = 0.0f;
    if (EPI == 1) biasv = biasPtr[0];

#pragma unroll
    for (int ti = 0; ti < 2; ++ti) {
#pragma unroll
        for (int tj = 0; tj < 2; ++tj) {
            const int r0 = rowBase + ti * 16 + h * 8;
            const int c0 = colBase + tj * 16 + n;
#pragma unroll
            for (int r = 0; r < 8; ++r) {
                float v = acc[ti][tj][r];
                const long row = r0 + r;
                if (EPI == 1)
                    v += s0v[(long)b * LL + row] + s1v[(long)b * LL + c0] + biasv;
                const long off = (long)b * sC + row * (long)ldc + c0;
                if (EPI == 2) Cb[off] = (__bf16)v;
                else          Cf[off] = v;
            }
        }
    }
}

// ---------------------------------------------------------------------------
// Pack kernel: cs = bf16(c * cq_weight);  s0[b,i] = c.c_weight;
// s1[b,i] = c.q_weight;  out region0 = c.   One block per (b, i) row.
// ---------------------------------------------------------------------------
__global__ __launch_bounds__(256) void pack_scale(
    const float* __restrict__ c, const float* __restrict__ cw,
    const float* __restrict__ qw, const float* __restrict__ cqw,
    __bf16* __restrict__ cs, float* __restrict__ outR0,
    float* __restrict__ s0, float* __restrict__ s1v) {
    const int bi = blockIdx.x;       // b*L + i
    const int d  = threadIdx.x;      // 0..255
    const long idx = (long)bi * DD + d;
    const float cv = c[idx];
    cs[idx]    = (__bf16)(cv * cqw[d]);
    outR0[idx] = cv;

    __shared__ float r0[256], r1[256];
    r0[d] = cv * cw[d];
    r1[d] = cv * qw[d];
    __syncthreads();
#pragma unroll
    for (int s = 128; s > 0; s >>= 1) {
        if (d < s) { r0[d] += r0[d + s]; r1[d] += r1[d + s]; }
        __syncthreads();
    }
    if (d == 0) { s0[bi] = r0[0]; s1v[bi] = r1[0]; }
}

// ---------------------------------------------------------------------------
// Tiled transpose + bf16 pack: X[b, L, D] -> Xrow bf16 [b, L, D] and
// XT bf16 [b, D, L].  32x32 tiles through LDS (+1 pad).
// ---------------------------------------------------------------------------
__global__ __launch_bounds__(256) void transpose_pack(
    const float* __restrict__ X, __bf16* __restrict__ Xrow,
    __bf16* __restrict__ XT) {
    const int b  = blockIdx.z;
    const int d0 = blockIdx.x * 32;  // over D
    const int i0 = blockIdx.y * 32;  // over L
    __shared__ float t[32][33];
#pragma unroll
    for (int r = 0; r < 4; ++r) {
        const int row = i0 + threadIdx.y + r * 8;
        const int col = d0 + threadIdx.x;
        const float v = X[((long)b * LL + row) * DD + col];
        t[threadIdx.y + r * 8][threadIdx.x] = v;
        Xrow[((long)b * LL + row) * DD + col] = (__bf16)v;
    }
    __syncthreads();
#pragma unroll
    for (int r = 0; r < 4; ++r) {
        const int drow = d0 + threadIdx.y + r * 8;
        const int icol = i0 + threadIdx.x;
        XT[((long)b * DD + drow) * LL + icol] =
            (__bf16)t[threadIdx.x][threadIdx.y + r * 8];
    }
}

// ---------------------------------------------------------------------------
// Row softmax over j (axis=-1) with mask_logits(x,m) = x*m + x*(1-m)*1e30.
// One block per (b, i) row of 1024; 4 elements per thread.  bf16 output.
// ---------------------------------------------------------------------------
__global__ __launch_bounds__(256) void row_softmax(
    const float* __restrict__ S, const float* __restrict__ qmask,
    __bf16* __restrict__ S1) {
    const long row  = blockIdx.x;         // b*L + i
    const int  b    = (int)(row >> 10);
    const long base = row << 10;
    const int  t    = threadIdx.x;

    float x[4];
    float m = -3.0e38f;
#pragma unroll
    for (int u = 0; u < 4; ++u) {
        const int j = t + u * 256;
        const float v  = S[base + j];
        const float qm = qmask[((long)b << 10) + j];
        const float ml = v * qm + v * (1.0f - qm) * 1e30f;
        x[u] = ml;
        m = fmaxf(m, ml);
    }
    __shared__ float red[256];
    red[t] = m;
    __syncthreads();
#pragma unroll
    for (int s = 128; s > 0; s >>= 1) {
        if (t < s) red[t] = fmaxf(red[t], red[t + s]);
        __syncthreads();
    }
    m = red[0];
    __syncthreads();

    float e[4];
    float sum = 0.0f;
#pragma unroll
    for (int u = 0; u < 4; ++u) { e[u] = __expf(x[u] - m); sum += e[u]; }
    red[t] = sum;
    __syncthreads();
#pragma unroll
    for (int s = 128; s > 0; s >>= 1) {
        if (t < s) red[t] += red[t + s];
        __syncthreads();
    }
    sum = red[0];
    const float inv = 1.0f / sum;
#pragma unroll
    for (int u = 0; u < 4; ++u)
        S1[base + t + u * 256] = (__bf16)(e[u] * inv);
}

// ---------------------------------------------------------------------------
// Batch-axis softmax (torch legacy dim=0 on swapaxes(S,1,2)) emitted directly
// in transposed bf16 layout for the NT GEMM:
//   S2T[b, n, j] = exp(ml(S[b,n,j], c_mask[b,n]) - m[n,j]) / sum_b exp(...)
// ---------------------------------------------------------------------------
__global__ __launch_bounds__(256) void batch_softmax(
    const float* __restrict__ S, const float* __restrict__ cmask,
    __bf16* __restrict__ S2T) {
    const long tid = (long)blockIdx.x * 256 + threadIdx.x;  // over L*L
    const int  n = (int)(tid >> 10);
    const int  j = (int)(tid & 1023);

    float xv[BB];
    float m = -3.0e38f;
#pragma unroll
    for (int b = 0; b < BB; ++b) {
        const long off = ((long)b * LL + n) * LL + j;
        const float v  = S[off];
        const float cm = cmask[(long)b * LL + n];
        const float ml = v * cm + v * (1.0f - cm) * 1e30f;
        xv[b] = ml;
        m = fmaxf(m, ml);
    }
    float s = 0.0f;
#pragma unroll
    for (int b = 0; b < BB; ++b) { xv[b] = __expf(xv[b] - m); s += xv[b]; }
    const float inv = 1.0f / s;
#pragma unroll
    for (int b = 0; b < BB; ++b)
        S2T[((long)b * LL + n) * LL + j] = (__bf16)(xv[b] * inv);
}

// ---------------------------------------------------------------------------
// Final elementwise: out[2B..] = c * C2Q ; out[3B..] = c * Q2C (in place).
// ---------------------------------------------------------------------------
__global__ __launch_bounds__(256) void final_mul(
    const float* __restrict__ c, float* __restrict__ out, long NLD) {
    const long idx = (long)blockIdx.x * 256 + threadIdx.x;
    const float cv = c[idx];
    out[2 * NLD + idx] = cv * out[NLD + idx];
    out[3 * NLD + idx] = cv * out[3 * NLD + idx];
}

// ---------------------------------------------------------------------------
extern "C" void kernel_launch(void* const* d_in, const int* in_sizes, int n_in,
                              void* d_out, int out_size, void* d_ws,
                              size_t ws_size, hipStream_t stream) {
    (void)in_sizes; (void)n_in; (void)out_size; (void)ws_size;

    const float* c      = (const float*)d_in[0];
    const float* q      = (const float*)d_in[1];
    const float* c_mask = (const float*)d_in[2];  // [B,1,L]
    const float* q_mask = (const float*)d_in[3];  // [B,1,L]
    const float* c_w    = (const float*)d_in[4];  // [D]
    const float* q_w    = (const float*)d_in[5];  // [D]
    const float* cq_w   = (const float*)d_in[6];  // [D]
    const float* bias   = (const float*)d_in[7];  // [1]
    float* out          = (float*)d_out;          // [4B, L, D]

    const long NLD = (long)BB * LL * DD;  // 4,194,304
    const long NLL = (long)BB * LL * LL;  // 16,777,216

    // Workspace carve-out (~200 MiB, all 16B-aligned chunks).
    char* w = (char*)d_ws;
    float*  S    = (float*)w;  w += NLL * sizeof(float);
    __bf16* S1   = (__bf16*)w; w += NLL * sizeof(__bf16);
    __bf16* S2T  = (__bf16*)w; w += NLL * sizeof(__bf16);
    __bf16* Tm   = (__bf16*)w; w += NLL * sizeof(__bf16);
    __bf16* cs   = (__bf16*)w; w += NLD * sizeof(__bf16);
    __bf16* qrow = (__bf16*)w; w += NLD * sizeof(__bf16);
    __bf16* crow = (__bf16*)w; w += NLD * sizeof(__bf16);
    __bf16* qT   = (__bf16*)w; w += NLD * sizeof(__bf16);
    __bf16* cT   = (__bf16*)w; w += NLD * sizeof(__bf16);
    float*  s0   = (float*)w;  w += (long)BB * LL * sizeof(float);
    float*  s1v  = (float*)w;  w += (long)BB * LL * sizeof(float);

    // 1) pack cs = bf16(c*cq_w), row/col bias terms, copy c -> out region 0.
    pack_scale<<<BB * LL, 256, 0, stream>>>(c, c_w, q_w, cq_w, cs, out, s0, s1v);

    // 2) bf16 packs + transposes of q and c.
    {
        dim3 blk(32, 8);
        dim3 grd(DD / 32, LL / 32, BB);
        transpose_pack<<<grd, blk, 0, stream>>>(q, qrow, qT);
        transpose_pack<<<grd, blk, 0, stream>>>(c, crow, cT);
    }

    // 3) S = cs . q^T + s0 + s1 + bias   (M=N=1024, K=256)
    gemm_nt<1><<<dim3(LL / 64, LL / 128, BB), 256, 0, stream>>>(
        cs, DD, NLD, qrow, DD, NLD, S, nullptr, LL, (long)LL * LL, DD,
        s0, s1v, bias);

    // 4) S1 = row softmax (bf16)
    row_softmax<<<BB * LL, 256, 0, stream>>>(S, q_mask, S1);

    // 5) S2T = batch-axis softmax, transposed layout (bf16)
    batch_softmax<<<(LL * LL) / 256, 256, 0, stream>>>(S, c_mask, S2T);

    // 6) C2Q = S1 . q   (NT against qT; M=1024, N=256, K=1024) -> out region 1
    gemm_nt<0><<<dim3(DD / 64, LL / 128, BB), 256, 0, stream>>>(
        S1, LL, NLL, qT, LL, NLD, out + NLD, nullptr, DD, NLD, LL,
        nullptr, nullptr, nullptr);

    // 7) T = S1 . S2   (NT against S2T; M=N=K=1024, bf16 out)
    gemm_nt<2><<<dim3(LL / 64, LL / 128, BB), 256, 0, stream>>>(
        S1, LL, NLL, S2T, LL, NLL, nullptr, Tm, LL, (long)LL * LL, LL,
        nullptr, nullptr, nullptr);

    // 8) Q2C = T . c   (NT against cT; M=1024, N=256, K=1024) -> out region 3
    gemm_nt<0><<<dim3(DD / 64, LL / 128, BB), 256, 0, stream>>>(
        Tm, LL, NLL, cT, LL, NLD, out + 3 * NLD, nullptr, DD, NLD, LL,
        nullptr, nullptr, nullptr);

    // 9) out region 2 = c * C2Q ; region 3 *= c
    final_mul<<<NLD / 256, 256, 0, stream>>>(c, out, NLD);
}
